// PyGRandLANet_326417514816
// MI455X (gfx1250) — matrix operand order repaired and checked by reading, not compiled, not run on hardware
//
#include <hip/hip_runtime.h>
#include <hip/hip_bf16.h>
#include <math.h>

typedef __attribute__((ext_vector_type(16))) _Float16 v16h;
typedef __attribute__((ext_vector_type(8)))  _Float16 v8h;
typedef __attribute__((ext_vector_type(4)))  _Float16 v4h;
typedef __attribute__((ext_vector_type(8)))  float    v8f;

#define DEC 4
#define KNN 16

// ---------------------------------------------------------------------------
// Wave32 reductions
// ---------------------------------------------------------------------------
static __device__ __forceinline__ float wave_max32(float v) {
#pragma unroll
  for (int o = 16; o > 0; o >>= 1) v = fmaxf(v, __shfl_xor(v, o, 32));
  return v;
}
static __device__ __forceinline__ float wave_sum32(float v) {
#pragma unroll
  for (int o = 16; o > 0; o >>= 1) v += __shfl_xor(v, o, 32);
  return v;
}

// ---------------------------------------------------------------------------
// GEMM: C[M,N] = act(A[M,K] @ W[K,N] + bias) via v_wmma_f32_16x16x32_f16.
// 128-thread workgroup = 4 waves, 64x64 C tile; each wave owns a 32x32
// sub-tile (2x2 WMMA), K staged 64-deep in LDS as f16.
// Fast path (full tile, aligned): float4 global loads -> packed LDS stores.
// Masked path: clamped-address loads made un-sinkable via multiply-mask.
// Fragment layouts used here (wave32, contiguous k):
//   A: lane half h, M = lane&15; a[j]: K = h*16 + j
//   B: N = lane&15;              b[j]: K = h*16 + j
//   C: c[r]: M = r + 8*h, N = lane&15
// slope < 0 -> no activation, else leaky_relu(slope).
// ---------------------------------------------------------------------------
#define GEMM_MT 64
#define GEMM_NT 64
#define GEMM_KT 64

__global__ __launch_bounds__(128)
void gemm_wmma(const float* __restrict__ A, const float* __restrict__ W,
               const float* __restrict__ bias, float* __restrict__ C,
               int M, int N, int K, float slope) {
  __shared__ _Float16 As[GEMM_MT][GEMM_KT];  // As[m][k]
  __shared__ _Float16 Bs[GEMM_NT][GEMM_KT];  // Bs[n][k]  (transposed)

  const int tid  = threadIdx.x;
  const int wave = tid >> 5;       // 0..3
  const int lane = tid & 31;
  const int half = lane >> 4;
  const int lm   = lane & 15;
  const int wm   = (wave >> 1) * 32;  // wave M offset in tile
  const int wn   = (wave & 1) * 32;   // wave N offset in tile
  const int row0 = blockIdx.y * GEMM_MT;
  const int col0 = blockIdx.x * GEMM_NT;

  const int mMax = M - 1, nMax = N - 1, kMax = K - 1;
  const bool fullMN = (row0 + GEMM_MT <= M) && (col0 + GEMM_NT <= N);
  const bool aligned = ((K & 3) == 0) && ((N & 3) == 0);

  v8f acc00 = {}, acc01 = {}, acc10 = {}, acc11 = {};

  for (int k0 = 0; k0 < K; k0 += GEMM_KT) {
    const bool fast = fullMN && aligned && (k0 + GEMM_KT <= K);
    if (fast) {
      // ---- vectorized staging: 1024 float4 chunks per matrix ----
#pragma unroll
      for (int i = 0; i < 8; ++i) {
        int c = tid + i * 128;                 // 0..1023
        int r = c >> 4, kk = (c & 15) << 2;    // A: row r, k-chunk kk
        float4 va = *(const float4*)&A[(size_t)(row0 + r) * K + (k0 + kk)];
        v4h ha = {(_Float16)va.x, (_Float16)va.y, (_Float16)va.z, (_Float16)va.w};
        *(v4h*)&As[r][kk] = ha;
      }
#pragma unroll
      for (int i = 0; i < 8; ++i) {
        int c = tid + i * 128;
        int kk = c >> 4, nn4 = (c & 15) << 2;  // B: k-row kk, n-chunk nn4
        float4 vb = *(const float4*)&W[(size_t)(k0 + kk) * N + (col0 + nn4)];
        Bs[nn4 + 0][kk] = (_Float16)vb.x;
        Bs[nn4 + 1][kk] = (_Float16)vb.y;
        Bs[nn4 + 2][kk] = (_Float16)vb.z;
        Bs[nn4 + 3][kk] = (_Float16)vb.w;
      }
    } else {
      // ---- masked staging, branch-free (multiply-mask keeps loads hoisted) ----
#pragma unroll
      for (int i = 0; i < 32; ++i) {
        int idx = tid + i * 128;               // 0..4095
        int r = idx >> 6, kk = idx & 63;
        int gr = row0 + r, gk = k0 + kk;
        int cr = gr < M ? gr : mMax;
        int ck = gk < K ? gk : kMax;
        float v = A[(size_t)cr * K + ck];
        float msk = (gr < M && gk < K) ? 1.0f : 0.0f;
        As[r][kk] = (_Float16)(v * msk);
      }
#pragma unroll
      for (int i = 0; i < 32; ++i) {
        int idx = tid + i * 128;
        int n = idx >> 6, kk = idx & 63;
        int gn = col0 + n, gk = k0 + kk;
        int cn = gn < N ? gn : nMax;
        int ck = gk < K ? gk : kMax;
        float v = W[(size_t)ck * N + cn];
        float msk = (gn < N && gk < K) ? 1.0f : 0.0f;
        Bs[n][kk] = (_Float16)(v * msk);
      }
    }
    __syncthreads();

    // ---- two 32-deep WMMA steps (8 static v_wmma) ----
#pragma unroll
    for (int sk = 0; sk < 2; ++sk) {
      const int kb = sk * 32;
      v16h a0 = *(const v16h*)&As[wm + lm][kb + (half << 4)];
      v16h a1 = *(const v16h*)&As[wm + 16 + lm][kb + (half << 4)];
      v16h b0 = *(const v16h*)&Bs[wn + lm][kb + (half << 4)];
      v16h b1 = *(const v16h*)&Bs[wn + 16 + lm][kb + (half << 4)];

      acc00 = __builtin_amdgcn_wmma_f32_16x16x32_f16(false, a0, false, b0, (short)0, acc00, false, false);
      acc01 = __builtin_amdgcn_wmma_f32_16x16x32_f16(false, a0, false, b1, (short)0, acc01, false, false);
      acc10 = __builtin_amdgcn_wmma_f32_16x16x32_f16(false, a1, false, b0, (short)0, acc10, false, false);
      acc11 = __builtin_amdgcn_wmma_f32_16x16x32_f16(false, a1, false, b1, (short)0, acc11, false, false);
    }
    __syncthreads();
  }

  // ---- epilogue ----
  v8f* accs[4] = {&acc00, &acc01, &acc10, &acc11};
  if (fullMN) {
#pragma unroll
    for (int tm = 0; tm < 2; ++tm) {
#pragma unroll
      for (int tn = 0; tn < 2; ++tn) {
        v8f& acc = *accs[tm * 2 + tn];
        const int nc = col0 + wn + tn * 16 + lm;
        const float bv = bias ? bias[nc] : 0.0f;
#pragma unroll
        for (int r = 0; r < 8; ++r) {
          int mr = row0 + wm + tm * 16 + r + (half << 3);
          float v = acc[r] + bv;
          if (slope >= 0.0f && v < 0.0f) v *= slope;
          C[(size_t)mr * N + nc] = v;
        }
      }
    }
  } else {
#pragma unroll
    for (int tm = 0; tm < 2; ++tm) {
#pragma unroll
      for (int tn = 0; tn < 2; ++tn) {
        v8f& acc = *accs[tm * 2 + tn];
        const int nc = col0 + wn + tn * 16 + lm;
        if (nc < N) {
          const float bv = bias ? bias[nc] : 0.0f;
#pragma unroll
          for (int r = 0; r < 8; ++r) {
            int mr = row0 + wm + tm * 16 + r + (half << 3);
            if (mr < M) {
              float v = acc[r] + bv;
              if (slope >= 0.0f && v < 0.0f) v *= slope;
              C[(size_t)mr * N + nc] = v;
            }
          }
        }
      }
    }
  }
}

// ---------------------------------------------------------------------------
// Brute-force KNN: one wave per query. Query q sits at pos[4*q] (decimation).
// Per-lane sorted top-16 insertion lists, then wave-cooperative 32-way merge.
// ---------------------------------------------------------------------------
__global__ __launch_bounds__(32)
void knn_kernel(const float* __restrict__ pos, int N, int* __restrict__ col) {
  __shared__ float sd[32 * KNN];
  __shared__ int   si[32 * KNN];
  const int q = blockIdx.x;
  const int lane = threadIdx.x;
  const int qi = q * DEC;
  const float qx = pos[3 * qi], qy = pos[3 * qi + 1], qz = pos[3 * qi + 2];
  float bd[KNN]; int bi[KNN];
#pragma unroll
  for (int t = 0; t < KNN; ++t) { bd[t] = 3.4e38f; bi[t] = 0; }
  for (int c = lane; c < N; c += 32) {
    if (c + 128 < N) __builtin_prefetch(&pos[3 * (c + 128)], 0, 0);
    float dx = pos[3 * c] - qx, dy = pos[3 * c + 1] - qy, dz = pos[3 * c + 2] - qz;
    float d = dx * dx + dy * dy + dz * dz;
    if (d < bd[KNN - 1]) {
      bd[KNN - 1] = d; bi[KNN - 1] = c;
#pragma unroll
      for (int t = KNN - 1; t > 0; --t) {
        if (bd[t] < bd[t - 1]) {
          float td = bd[t]; bd[t] = bd[t - 1]; bd[t - 1] = td;
          int   ti = bi[t]; bi[t] = bi[t - 1]; bi[t - 1] = ti;
        }
      }
    }
  }
#pragma unroll
  for (int t = 0; t < KNN; ++t) { sd[lane * KNN + t] = bd[t]; si[lane * KNN + t] = bi[t]; }
  __syncthreads();
  int head = 0;
  for (int k = 0; k < KNN; ++k) {
    float v = (head < KNN) ? sd[lane * KNN + head] : 3.4e38f;
    float bv = v; int bl = lane;
#pragma unroll
    for (int o = 16; o > 0; o >>= 1) {
      float ov = __shfl_xor(bv, o, 32);
      int   ol = __shfl_xor(bl, o, 32);
      if (ov < bv || (ov == bv && ol < bl)) { bv = ov; bl = ol; }
    }
    int bh = __shfl(head, bl, 32);
    if (lane == 0) col[(size_t)q * KNN + k] = si[bl * KNN + bh];
    if (lane == bl) head++;
  }
}

// ---------------------------------------------------------------------------
// Edge positional encoding: lse[e,f] = rel(e) . encW[:,f] + encb[f]
// rel = [pos_i(3), pos_j(3), d(3), L1(d)] with pos_i = pos[row] (ref quirk).
// ---------------------------------------------------------------------------
__global__ void edge_encode_kernel(const float* __restrict__ pos, const int* __restrict__ col,
                                   const float* __restrict__ eW, const float* __restrict__ eb,
                                   float* __restrict__ lse, int E, int f) {
  long long t = (long long)blockIdx.x * blockDim.x + threadIdx.x;
  if (t >= (long long)E * f) return;
  int e = (int)(t / f), o = (int)(t % f);
  int r = e / KNN, c = col[e];
  float pix = pos[3 * r], piy = pos[3 * r + 1], piz = pos[3 * r + 2];
  float pjx = pos[3 * c], pjy = pos[3 * c + 1], pjz = pos[3 * c + 2];
  float dx = pjx - pix, dy = pjy - piy, dz = pjz - piz;
  float eu = fabsf(dx) + fabsf(dy) + fabsf(dz);
  float rel[10] = {pix, piy, piz, pjx, pjy, pjz, dx, dy, dz, eu};
  float acc = eb[o];
#pragma unroll
  for (int k = 0; k < 10; ++k) acc += rel[k] * eW[k * f + o];
  lse[t] = acc;
}

// out[i] = [ A[idx[i]] (da feats) , B[i] (db feats) ]
__global__ void gather_concat_kernel(const float* __restrict__ A, const int* __restrict__ idx,
                                     const float* __restrict__ B, float* __restrict__ o,
                                     int n, int da, int db) {
  const int tot = da + db;
  long long t = (long long)blockIdx.x * blockDim.x + threadIdx.x;
  if (t >= (long long)n * tot) return;
  int i = (int)(t / tot), f = (int)(t % tot);
  o[t] = (f < da) ? A[(size_t)idx[i] * da + f] : B[(size_t)i * db + (f - da)];
}

// Per-edge softmax over F features, multiply by out1, accumulate into row e/16.
__global__ __launch_bounds__(32)
void softmax_scatter_kernel(const float* __restrict__ logits, const float* __restrict__ o1,
                            float* __restrict__ outp, int F) {
  const int e = blockIdx.x;
  const int lane = threadIdx.x;
  const int r = e / KNN;
  const float* L = logits + (size_t)e * F;
  const float* O = o1 + (size_t)e * F;
  float* dst = outp + (size_t)r * F;
  float mx = -3.4e38f;
  for (int f = lane; f < F; f += 32) mx = fmaxf(mx, L[f]);
  mx = wave_max32(mx);
  float sum = 0.0f;
  for (int f = lane; f < F; f += 32) sum += __expf(L[f] - mx);
  sum = wave_sum32(sum);
  const float inv = 1.0f / sum;
  for (int f = lane; f < F; f += 32)
    atomicAdd(&dst[f], __expf(L[f] - mx) * inv * O[f]);
}

__global__ void zero_kernel(float* __restrict__ p, long long n) {
  long long t = (long long)blockIdx.x * blockDim.x + threadIdx.x;
  if (t < n) p[t] = 0.0f;
}

// o = leaky(a + b, slope)
__global__ void add_leaky_kernel(const float* __restrict__ a, const float* __restrict__ b,
                                 float* __restrict__ o, long long n, float slope) {
  long long t = (long long)blockIdx.x * blockDim.x + threadIdx.x;
  if (t >= n) return;
  float v = a[t] + b[t];
  o[t] = v > 0.0f ? v : v * slope;
}

// o[i,:] = A[i*stride,:]   (decimation gather)
__global__ void downsample_kernel(const float* __restrict__ A, float* __restrict__ o,
                                  int nq, int f, int stride) {
  long long t = (long long)blockIdx.x * blockDim.x + threadIdx.x;
  if (t >= (long long)nq * f) return;
  int i = (int)(t / f), c = (int)(t % f);
  o[t] = A[(size_t)(i * stride) * f + c];
}

// nn[i] = argmin_j ||ps[i]-psrc[j]||^2
__global__ void nn1_kernel(const float* __restrict__ ps, const float* __restrict__ psrc,
                           int n, int m, int* __restrict__ nn) {
  int i = blockIdx.x * blockDim.x + threadIdx.x;
  if (i >= n) return;
  float px = ps[3 * i], py = ps[3 * i + 1], pz = ps[3 * i + 2];
  float best = 3.4e38f; int bj = 0;
  for (int j = 0; j < m; ++j) {
    if (j + 32 < m) __builtin_prefetch(&psrc[3 * (j + 32)], 0, 0);
    float dx = psrc[3 * j] - px, dy = psrc[3 * j + 1] - py, dz = psrc[3 * j + 2] - pz;
    float d = dx * dx + dy * dy + dz * dz;
    if (d < best) { best = d; bj = j; }
  }
  nn[i] = bj;
}

// BatchNorm1d (eval) + ReLU, in place
__global__ void bn_relu_kernel(float* __restrict__ z, const float* __restrict__ g,
                               const float* __restrict__ bt, const float* __restrict__ mu,
                               const float* __restrict__ var, long long n, int f) {
  long long t = (long long)blockIdx.x * blockDim.x + threadIdx.x;
  if (t >= n) return;
  int c = (int)(t % f);
  float v = (z[t] - mu[c]) * rsqrtf(var[c] + 1e-5f) * g[c] + bt[c];
  z[t] = v > 0.0f ? v : 0.0f;
}

// ---------------------------------------------------------------------------
// Host orchestration
// ---------------------------------------------------------------------------
extern "C" void kernel_launch(void* const* d_in, const int* in_sizes, int n_in,
                              void* d_out, int out_size, void* d_ws, size_t ws_size,
                              hipStream_t stream) {
  (void)in_sizes; (void)n_in; (void)out_size; (void)ws_size;
  const float* x   = (const float*)d_in[0];
  const float* pos = (const float*)d_in[1];
  // d_in[2] = batch (all zeros, unused)
  auto P = [&](int i) { return (const float*)d_in[i]; };

  const int din_[4] = {6, 32, 128, 256};
  const int dou_[4] = {32, 128, 256, 512};
  const int Np[5]   = {16384, 4096, 1024, 256, 64};

  float* wsf = (float*)d_ws;
  size_t off = 0;
  auto alloc = [&](size_t n) { float* p = wsf + off; off += (n + 63) & ~(size_t)63; return p; };

  auto gemm = [&](const float* A, const float* Wp, const float* bp, float* Cp,
                  int M, int N, int K, float slope) {
    dim3 g((N + GEMM_NT - 1) / GEMM_NT, (M + GEMM_MT - 1) / GEMM_MT);
    gemm_wmma<<<g, dim3(128), 0, stream>>>(A, Wp, bp, Cp, M, N, K, slope);
  };
  auto nblk = [](long long n) { return (unsigned)((n + 255) / 256); };

  // persistent per-block outputs
  float* hb[4]; float* pb[4];
  for (int b = 0; b < 4; ++b) {
    hb[b] = alloc((size_t)Np[b + 1] * dou_[b]);
    pb[b] = alloc((size_t)Np[b + 1] * 3);
  }
  const size_t mark = off;

  const float* hin = x;
  const float* pin = pos;
  for (int b = 0; b < 4; ++b) {
    off = mark;
    const int N = Np[b], Nq = Np[b + 1], E = Nq * KNN;
    const int di = din_[b], d = dou_[b];
    const int base = 3 + 14 * b;  // sorted keys: l1a,l1e,l2a,l2e,m1,m2,sc ; (W,b) each
    const float *l1aW = P(base + 0),  *l1ab = P(base + 1);
    const float *l1eW = P(base + 2),  *l1eb = P(base + 3);
    const float *l2aW = P(base + 4),  *l2ab = P(base + 5);
    const float *l2eW = P(base + 6),  *l2eb = P(base + 7);
    const float *m1W  = P(base + 8),  *m1b  = P(base + 9);
    const float *m2W  = P(base + 10), *m2b  = P(base + 11);
    const float *scW  = P(base + 12), *scb  = P(base + 13);

    int*   col  = (int*)alloc(E);
    float* sc   = alloc((size_t)N * d);
    float* h1   = alloc((size_t)N * (d / 4));
    float* lse1 = alloc((size_t)E * (d / 4));
    float* o1   = alloc((size_t)E * (d / 2));
    float* lg1  = alloc((size_t)E * (d / 2));
    float* hl1  = alloc((size_t)N * (d / 2));
    float* lse2 = alloc((size_t)E * (d / 2));
    float* o2   = alloc((size_t)E * d);
    float* lg2  = alloc((size_t)E * d);
    float* hl2  = alloc((size_t)N * d);
    float* t2   = alloc((size_t)N * d);

    knn_kernel<<<Nq, 32, 0, stream>>>(pin, N, col);
    gemm(hin, scW, scb, sc, N, d, di, 0.2f);          // shortcut
    gemm(hin, m1W, m1b, h1, N, d / 4, di, 0.2f);      // mlp1
    // LFA 1 (d/2)
    edge_encode_kernel<<<nblk((long long)E * (d / 4)), 256, 0, stream>>>(pin, col, l1eW, l1eb, lse1, E, d / 4);
    gather_concat_kernel<<<nblk((long long)E * (d / 2)), 256, 0, stream>>>(h1, col, lse1, o1, E, d / 4, d / 4);
    gemm(o1, l1aW, l1ab, lg1, E, d / 2, d / 2, -1.0f);
    zero_kernel<<<nblk((long long)N * (d / 2)), 256, 0, stream>>>(hl1, (long long)N * (d / 2));
    softmax_scatter_kernel<<<E, 32, 0, stream>>>(lg1, o1, hl1, d / 2);
    // LFA 2 (d)
    edge_encode_kernel<<<nblk((long long)E * (d / 2)), 256, 0, stream>>>(pin, col, l2eW, l2eb, lse2, E, d / 2);
    gather_concat_kernel<<<nblk((long long)E * d), 256, 0, stream>>>(hl1, col, lse2, o2, E, d / 2, d / 2);
    gemm(o2, l2aW, l2ab, lg2, E, d, d, -1.0f);
    zero_kernel<<<nblk((long long)N * d), 256, 0, stream>>>(hl2, (long long)N * d);
    softmax_scatter_kernel<<<E, 32, 0, stream>>>(lg2, o2, hl2, d);
    // mlp2 + residual
    gemm(hl2, m2W, m2b, t2, N, d, d, 0.2f);
    add_leaky_kernel<<<nblk((long long)N * d), 256, 0, stream>>>(t2, sc, t2, (long long)N * d, 0.01f);
    // decimate
    downsample_kernel<<<nblk((long long)Nq * d), 256, 0, stream>>>(t2, hb[b], Nq, d, DEC);
    downsample_kernel<<<nblk((long long)Nq * 3), 256, 0, stream>>>(pin, pb[b], Nq, 3, DEC);
    hin = hb[b]; pin = pb[b];
  }

  // -------- decoder --------
  off = mark;
  const float *bng = P(59), *bnb = P(60), *bnmu = P(61), *bnvar = P(62);
  const float *fpW[4] = {P(63), P(65), P(67), P(69)};
  const float *fpB[4] = {P(64), P(66), P(68), P(70)};
  const float *h1W = P(71), *h1b = P(72);
  const float *h2W = P(73), *h2b = P(74);
  const float *midW = P(75), *midb = P(76);
  const float *outW = P(77), *outb = P(78);

  float* hmid = alloc((size_t)64 * 512);
  gemm(hb[3], midW, midb, hmid, 64, 512, 512, -1.0f);

  struct Skip { const float* xs; const float* ps; int n; int f; };
  const Skip skips[4] = {{hb[2], pb[2], 256, 256}, {hb[1], pb[1], 1024, 128},
                         {hb[0], pb[0], 4096, 32}, {x, pos, 16384, 6}};
  const int fpout[4] = {256, 128, 32, 8};

  const float* hc = hmid; int hcf = 512;
  const float* psrc = pb[3]; int msrc = 64;
  for (int s = 0; s < 4; ++s) {
    const int n = skips[s].n, xf = skips[s].f;
    int*   nn = (int*)alloc(n);
    float* cc = alloc((size_t)n * (hcf + xf));
    float* ho = alloc((size_t)n * fpout[s]);
    nn1_kernel<<<nblk(n), 256, 0, stream>>>(skips[s].ps, psrc, n, msrc, nn);
    gather_concat_kernel<<<nblk((long long)n * (hcf + xf)), 256, 0, stream>>>(hc, nn, skips[s].xs, cc, n, hcf, xf);
    gemm(cc, fpW[s], fpB[s], ho, n, fpout[s], hcf + xf, -1.0f);
    hc = ho; hcf = fpout[s]; psrc = skips[s].ps; msrc = n;
  }

  // -------- head --------
  float* z1 = alloc((size_t)16384 * 64);
  gemm(hc, h1W, h1b, z1, 16384, 64, 8, -1.0f);
  bn_relu_kernel<<<nblk(16384LL * 64), 256, 0, stream>>>(z1, bng, bnb, bnmu, bnvar, 16384LL * 64, 64);
  float* z2 = alloc((size_t)16384 * 32);
  gemm(z1, h2W, h2b, z2, 16384, 32, 64, -1.0f);
  gemm(z2, outW, outb, (float*)d_out, 16384, 7, 32, -1.0f);
}
